// VectorQuantizer_61211873902974
// MI455X (gfx1250) — matrix operand order repaired and checked by reading, not compiled, not run on hardware
//
#include <hip/hip_runtime.h>

typedef float  v4f   __attribute__((ext_vector_type(4)));
typedef float  v8f   __attribute__((ext_vector_type(8)));
typedef __bf16 v16bf __attribute__((ext_vector_type(16)));

namespace {
constexpr int kN = 262144;          // rows
constexpr int kD = 256;             // dim
constexpr int kK = 640;             // codes
constexpr int kTiles = kK / 16;     // 40 column tiles
constexpr int kThreads = 256;       // 8 wave32
constexpr int kRowsPerBlock = 128;  // 8 waves x 16 rows
// packed-B layout: per (tile, kb, h/l): 32 lane slots of 24 bf16 (48 B, 32 B used)
constexpr int kLaneSlot = 24;                 // bf16 per lane slot
constexpr int kFragBlk  = 32 * kLaneSlot;     // 768 bf16 per (kb, hl)
constexpr int kTileSpan = 8 * 2 * kFragBlk;   // 12288 bf16 = 24 KB per tile
}

__device__ __forceinline__ void code_range(long long at, int& s, int& e) {
    if (at == 5)        { s = 0;   e = 128; }
    else if (at == 6)   { s = 129; e = 256; }
    else if (at == 7)   { s = 257; e = 384; }
    else if (at == 119) { s = 513; e = 576; }
    else if (at == 120) { s = 577; e = 640; }
    else                { s = 385; e = 512; }
}

__device__ __forceinline__ v8f wmma_bf16(v16bf a, v16bf b, v8f c) {
    return __builtin_amdgcn_wmma_f32_16x16x32_bf16(false, a, false, b,
                                                   (short)0, c, false, false);
}

// ---------------- prep kernels ----------------

__global__ void vq_init(double* acc) { *acc = 0.0; }

// w2[k] = sum_j W[k][j]^2  (exact f32)
__global__ __launch_bounds__(256) void vq_w2(const float* __restrict__ W,
                                             float* __restrict__ w2) {
    __shared__ float red[8];
    const int k = blockIdx.x;
    float v = W[(size_t)k * kD + threadIdx.x];
    float p = v * v;
    #pragma unroll
    for (int off = 16; off > 0; off >>= 1) p += __shfl_xor(p, off, 32);
    const int lane = threadIdx.x & 31, w = threadIdx.x >> 5;
    if (lane == 0) red[w] = p;
    __syncthreads();
    if (threadIdx.x == 0) {
        float s = 0.f;
        #pragma unroll
        for (int i = 0; i < 8; ++i) s += red[i];
        w2[k] = s;
    }
}

// Split W into bf16 hi/lo B-fragments, fragment order for V_WMMA_F32_16X16X32_BF16.
// B layout: lane n = l%16, elem j -> K = kb*32 + (j/8)*16 + (l/16)*8 + j%8.
__global__ __launch_bounds__(256) void vq_packb(const float* __restrict__ W,
                                                __bf16* __restrict__ packB) {
    const int t = blockIdx.x;
    for (int i = threadIdx.x; i < 8 * 2 * 32 * 16; i += 256) {
        const int kb   = i >> 10;
        const int rem  = i & 1023;
        const int hl   = rem >> 9;
        const int rem2 = rem & 511;
        const int l    = rem2 >> 4;
        const int j    = rem2 & 15;
        const int col  = kb * 32 + ((j >> 3) << 4) + (l >> 4) * 8 + (j & 7);
        const int row  = t * 16 + (l & 15);
        const float v  = W[(size_t)row * kD + col];
        const __bf16 h = (__bf16)v;
        const __bf16 lo = (__bf16)(v - (float)h);
        packB[(size_t)t * kTileSpan + (kb * 2 + hl) * kFragBlk + l * kLaneSlot + j] =
            hl ? lo : h;
    }
}

// ---------------- main kernel ----------------

__global__ __launch_bounds__(kThreads) void vq_main(
    const long long* __restrict__ x, const float* __restrict__ e,
    const float* __restrict__ W, const float* __restrict__ w2,
    const __bf16* __restrict__ packB, float* __restrict__ out,
    double* __restrict__ acc) {
    __shared__ alignas(16) __bf16 Bs[2][kTileSpan];  // 2 x 24 KB
    __shared__ float e2_s[kRowsPerBlock];
    __shared__ int   idx_s[kRowsPerBlock];
    __shared__ float red_s[kThreads / 32];

    const int tid = threadIdx.x;
    const int rowBase = blockIdx.x * kRowsPerBlock;
    const int wave   = tid >> 5;
    const int lane   = tid & 31;
    const int laneLo = lane & 15;
    const int half   = lane >> 4;
    const int strip0 = wave * 16;

    // ---- Build A fragments (bf16 hi/lo) in registers; accumulate ||e||^2 ----
    // A layout: lane row = laneLo; elems 0..7 -> K = kb*32 + half*8 + j,
    //           elems 8..15 -> K = kb*32 + 16 + half*8 + j.
    v16bf ah[8], al[8];
    float e2part = 0.f;
    {
        const float* rowPtr = e + (size_t)(rowBase + strip0 + laneLo) * kD;
        #pragma unroll
        for (int kb = 0; kb < 8; ++kb) {
            const int c0 = kb * 32 + half * 8;
            const v4f a0 = *(const v4f*)(rowPtr + c0);
            const v4f a1 = *(const v4f*)(rowPtr + c0 + 4);
            const v4f a2 = *(const v4f*)(rowPtr + c0 + 16);
            const v4f a3 = *(const v4f*)(rowPtr + c0 + 20);
            #pragma unroll
            for (int j = 0; j < 4; ++j) {
                const float v0 = a0[j], v1 = a1[j], v2 = a2[j], v3 = a3[j];
                e2part += v0 * v0 + v1 * v1 + v2 * v2 + v3 * v3;
                const __bf16 h0 = (__bf16)v0, h1 = (__bf16)v1;
                const __bf16 h2 = (__bf16)v2, h3 = (__bf16)v3;
                ah[kb][j]      = h0; al[kb][j]      = (__bf16)(v0 - (float)h0);
                ah[kb][4 + j]  = h1; al[kb][4 + j]  = (__bf16)(v1 - (float)h1);
                ah[kb][8 + j]  = h2; al[kb][8 + j]  = (__bf16)(v2 - (float)h2);
                ah[kb][12 + j] = h3; al[kb][12 + j] = (__bf16)(v3 - (float)h3);
            }
        }
    }
    // full row ||e||^2 lives on lanes 0..15 (row = lane)
    const float e2full = e2part + __shfl_xor(e2part, 16, 32);
    if (lane < 16) e2_s[strip0 + lane] = e2full;

    // ---- per-accumulator-slot code ranges (acc slot r -> row strip0+half*8+r) ----
    int sArr[8], eArr[8];
    #pragma unroll
    for (int r = 0; r < 8; ++r) {
        const long long at = x[(size_t)(rowBase + strip0 + half * 8 + r) * 2];
        code_range(at, sArr[r], eArr[r]);
    }

    float minv[8];
    int   mini[8];
    #pragma unroll
    for (int r = 0; r < 8; ++r) { minv[r] = 3.4e38f; mini[r] = 0; }

    // ---- stage tile 0 ----
    {
        const uint4* src = (const uint4*)(packB);
        uint4* dst = (uint4*)&Bs[0][0];
        for (int i = tid; i < kTileSpan / 8; i += kThreads) dst[i] = src[i];
    }
    __syncthreads();

    // ---- tile loop: 24 WMMAs per tile, double-buffered B in LDS ----
    for (int t = 0; t < kTiles; ++t) {
        const int cur = t & 1;
        if (t + 1 < kTiles) {
            const uint4* src = (const uint4*)(packB + (size_t)(t + 1) * kTileSpan);
            uint4* dst = (uint4*)&Bs[cur ^ 1][0];
            for (int i = tid; i < kTileSpan / 8; i += kThreads) dst[i] = src[i];
        }

        v8f c0v = {0.f, 0.f, 0.f, 0.f, 0.f, 0.f, 0.f, 0.f};
        v8f c1v = {0.f, 0.f, 0.f, 0.f, 0.f, 0.f, 0.f, 0.f};
        const __bf16* B = &Bs[cur][0];
        #pragma unroll
        for (int kb = 0; kb < 8; ++kb) {
            const v16bf bh = *(const v16bf*)(B + (kb * 2 + 0) * kFragBlk + lane * kLaneSlot);
            const v16bf bl = *(const v16bf*)(B + (kb * 2 + 1) * kFragBlk + lane * kLaneSlot);
            if ((kb & 1) == 0) {
                c0v = wmma_bf16(ah[kb], bh, c0v);
                c0v = wmma_bf16(al[kb], bh, c0v);
                c0v = wmma_bf16(ah[kb], bl, c0v);
            } else {
                c1v = wmma_bf16(ah[kb], bh, c1v);
                c1v = wmma_bf16(al[kb], bh, c1v);
                c1v = wmma_bf16(ah[kb], bl, c1v);
            }
        }

        const int   code = t * 16 + laneLo;
        const float w2v  = w2[code];
        #pragma unroll
        for (int r = 0; r < 8; ++r) {
            const float dot = c0v[r] + c1v[r];
            const float d2  = w2v - 2.0f * dot;  // ||e||^2 dropped for argmin
            const bool ok = (code >= sArr[r]) && (code < eArr[r]);
            if (ok && d2 < minv[r]) { minv[r] = d2; mini[r] = code; }
        }
        __syncthreads();
    }

    // ---- argmin across the 16 lanes sharing each row ----
    #pragma unroll
    for (int off = 1; off < 16; off <<= 1) {
        #pragma unroll
        for (int r = 0; r < 8; ++r) {
            const float ov = __shfl_xor(minv[r], off, 32);
            const int   oi = __shfl_xor(mini[r], off, 32);
            if (ov < minv[r] || (ov == minv[r] && oi < mini[r])) {
                minv[r] = ov; mini[r] = oi;
            }
        }
    }

    // ---- per-row loss = ||e||^2 + minv ; record idx ----
    float partial = 0.f;
    if (laneLo == 0) {
        #pragma unroll
        for (int r = 0; r < 8; ++r) {
            const int row = strip0 + half * 8 + r;
            idx_s[row] = mini[r];
            partial += e2_s[row] + minv[r];
        }
    }
    #pragma unroll
    for (int off = 16; off > 0; off >>= 1) partial += __shfl_xor(partial, off, 32);
    if (lane == 0) red_s[wave] = partial;
    __syncthreads();
    if (tid == 0) {
        float s = 0.f;
        #pragma unroll
        for (int i = 0; i < 8; ++i) s += red_s[i];
        atomicAdd(acc, (double)s);
    }

    // ---- out[row] = W[idx]; one full row per iteration, coalesced ----
    for (int i = tid; i < kRowsPerBlock * kD; i += kThreads) {
        const int r = i >> 8;
        const int c = i & 255;
        out[(size_t)(rowBase + r) * kD + c] = W[(size_t)idx_s[r] * kD + c];
    }
}

// ---------------- finalize ----------------

__global__ void vq_finalize(const double* __restrict__ acc,
                            float* __restrict__ tail) {
    const double m = *acc / ((double)kN * (double)kD);
    tail[0] = (float)m;          // codebook_loss
    tail[1] = (float)m;          // gnn_loss (identical forward value)
    tail[2] = (float)(1.25 * m); // vq_loss
}

// ---------------- launch ----------------

extern "C" void kernel_launch(void* const* d_in, const int* in_sizes, int n_in,
                              void* d_out, int out_size, void* d_ws, size_t ws_size,
                              hipStream_t stream) {
    const long long* x = (const long long*)d_in[0];  // int64 (N,2)
    const float*     e = (const float*)d_in[1];      // f32 (N,D)
    const float*     W = (const float*)d_in[2];      // f32 (K,D)
    float* out = (float*)d_out;                      // N*D quantized + 3 losses

    double* acc   = (double*)d_ws;                        // 8 B
    float*  w2    = (float*)((char*)d_ws + 256);          // 640 f32
    __bf16* packB = (__bf16*)((char*)d_ws + 4096);        // 40*24 KB = 960 KB

    vq_init<<<1, 1, 0, stream>>>(acc);
    vq_w2<<<kK, 256, 0, stream>>>(W, w2);
    vq_packb<<<kTiles, 256, 0, stream>>>(W, packB);
    vq_main<<<kN / kRowsPerBlock, kThreads, 0, stream>>>(x, e, W, w2, packB, out, acc);
    vq_finalize<<<1, 1, 0, stream>>>(acc, out + (size_t)kN * kD);
}
